// TransformerEncoder_22728966930443
// MI455X (gfx1250) — matrix operand order repaired and checked by reading, not compiled, not run on hardware
//
#include <hip/hip_runtime.h>
#include <hip/hip_bf16.h>
#include <math.h>

// ---------------------------------------------------------------------------
// CDNA5 / gfx1250 transformer encoder layer.
// All matmuls via v_wmma_f32_16x16x32_bf16 (bf16 inputs, fp32 accumulate).
// FC2 additionally stages its weight tile through LDS with
// global_load_async_to_lds_b128 / s_wait_asynccnt (double buffered).
// ---------------------------------------------------------------------------

typedef __bf16 bf16;
typedef __bf16 v16bf __attribute__((ext_vector_type(16)));
typedef __bf16 v8bf  __attribute__((ext_vector_type(8)));
typedef float  v8f   __attribute__((ext_vector_type(8)));
typedef int    v4i   __attribute__((ext_vector_type(4)));

#define AS1 __attribute__((address_space(1)))
#define AS3 __attribute__((address_space(3)))

#define BATCH  256
#define SEQ    80
#define DIM    1024
#define HEADS  8
#define HDIM   128           // DIM / HEADS
#define HID    4096
#define NTOK   (BATCH * SEQ) // 20480
#define SPAD   96            // SEQ padded to a multiple of 32 for WMMA K-dim
#define NBH    (BATCH * HEADS)

#if __has_builtin(__builtin_amdgcn_global_load_async_to_lds_b128) && \
    __has_builtin(__builtin_amdgcn_s_wait_asynccnt)
#define HAVE_ASYNC_LDS 1
#endif

__device__ __forceinline__ v16bf cat8(v8bf a, v8bf b) {
  return __builtin_shufflevector(a, b, 0,1,2,3,4,5,6,7,8,9,10,11,12,13,14,15);
}

__device__ __forceinline__ v8f wmma_bf16(v16bf a, v16bf b, v8f c) {
  // D(f32,16x16) = A(16x32 bf16) * B(32x16 bf16) + C
  return __builtin_amdgcn_wmma_f32_16x16x32_bf16(false, a, false, b,
                                                 (short)0, c, false, false);
}

// Wave-level GEMM: MI x NI grid of 16x16 f32 tiles, K-dim stepped by 32.
// A   : row-major [*, lda] bf16 (activations)
// Bt  : row-major [N, ldb] bf16 (pre-transposed weights, ldb == K)
template <int MI, int NI>
__device__ __forceinline__ void wave_gemm(const bf16* __restrict__ A, int lda,
                                          const bf16* __restrict__ Bt, int ldb,
                                          int K, int m0, int n0,
                                          v8f acc[MI][NI]) {
  const int lane = threadIdx.x & 31;
  const int hf = lane >> 4;   // wave half (ISA fragment layout)
  const int lr = lane & 15;
  for (int k0 = 0; k0 < K; k0 += 32) {
    v16bf a[MI];
#pragma unroll
    for (int mi = 0; mi < MI; ++mi) {
      const bf16* ap = A + (size_t)(m0 + mi * 16 + lr) * lda + k0 + 8 * hf;
      a[mi] = cat8(*(const v8bf*)ap, *(const v8bf*)(ap + 16));
    }
    v16bf b[NI];
#pragma unroll
    for (int ni = 0; ni < NI; ++ni) {
      const bf16* bp = Bt + (size_t)(n0 + ni * 16 + lr) * ldb + k0 + 8 * hf;
      b[ni] = cat8(*(const v8bf*)bp, *(const v8bf*)(bp + 16));
    }
    // Speculative next-slice prefetch (unconditional: OOB prefetch is dropped
    // silently per ISA 10.5, and this avoids a per-iteration branch).
    __builtin_prefetch(A + (size_t)(m0 + lr) * lda + k0 + 32, 0, 1);
    __builtin_prefetch(Bt + (size_t)(n0 + lr) * ldb + k0 + 32, 0, 1);
#pragma unroll
    for (int mi = 0; mi < MI; ++mi)
#pragma unroll
      for (int ni = 0; ni < NI; ++ni)
        acc[mi][ni] = wmma_bf16(a[mi], b[ni], acc[mi][ni]);
  }
}

// ---------------------------------------------------------------------------
// Prep kernels
// ---------------------------------------------------------------------------
__global__ void cast_kernel(const float* __restrict__ src,
                            bf16* __restrict__ dst, size_t n) {
  size_t i = (size_t)blockIdx.x * blockDim.x + threadIdx.x;
  if (i < n) dst[i] = (bf16)src[i];
}

// dst[n][k] = (bf16) src[k][n];  src is [K][N] f32, dst is [N][K] bf16
__global__ void transpose_cast_kernel(const float* __restrict__ src,
                                      bf16* __restrict__ dst, int K, int N) {
  size_t i = (size_t)blockIdx.x * blockDim.x + threadIdx.x;
  if (i >= (size_t)K * N) return;
  int n = (int)(i / K);
  int k = (int)(i % K);
  dst[i] = (bf16)src[(size_t)k * N + n];
}

__global__ void zero_kernel(uint4* __restrict__ dst, size_t n16) {
  size_t i = (size_t)blockIdx.x * blockDim.x + threadIdx.x;
  if (i < n16) dst[i] = make_uint4(0u, 0u, 0u, 0u);
}

// ---------------------------------------------------------------------------
// QKV projection: [20480 x 1024] @ [1024 x 3072] -> scattered q / k / v^T
// 8 waves, 64x64 per wave, 128x256 block tile.
// ---------------------------------------------------------------------------
__global__ __launch_bounds__(256) void gemm_qkv_kernel(
    const bf16* __restrict__ xb, const bf16* __restrict__ wqkvT,
    bf16* __restrict__ qb, bf16* __restrict__ kb, bf16* __restrict__ vtb) {
  const int w = threadIdx.x >> 5;
  const int wm = w & 1, wn = w >> 1;
  const int m0 = blockIdx.x * 128 + wm * 64;
  const int n0 = blockIdx.y * 256 + wn * 64;
  v8f acc[4][4] = {};
  wave_gemm<4, 4>(xb, DIM, wqkvT, DIM, DIM, m0, n0, acc);

  const int lane = threadIdx.x & 31, hf = lane >> 4, lr = lane & 15;
  const int region = (blockIdx.y * 256) >> 10;  // 0=q 1=k 2=v (block-uniform)
#pragma unroll
  for (int mi = 0; mi < 4; ++mi)
#pragma unroll
    for (int ni = 0; ni < 4; ++ni)
#pragma unroll
      for (int r = 0; r < 8; ++r) {
        int m = m0 + mi * 16 + 8 * hf + r;
        int n = n0 + ni * 16 + lr - (region << 10);
        int h = n >> 7, d = n & 127;
        int bidx = m / SEQ, s = m - bidx * SEQ;
        int bh = bidx * HEADS + h;
        bf16 val = (bf16)acc[mi][ni][r];
        if (region == 0)
          qb[((size_t)bh * SEQ + s) * HDIM + d] = val;
        else if (region == 1)
          kb[((size_t)bh * SEQ + s) * HDIM + d] = val;
        else
          vtb[((size_t)bh * HDIM + d) * SPAD + s] = val;  // V transposed
      }
}

// ---------------------------------------------------------------------------
// Attention: per (batch,head): QK^T -> softmax -> +bias -> P@V
// 5 waves, one 16-query M-tile each.
// ---------------------------------------------------------------------------
__global__ __launch_bounds__(160) void attn_kernel(
    const bf16* __restrict__ qb, const bf16* __restrict__ kb,
    const bf16* __restrict__ vtb, const float* __restrict__ bias_table,
    float* __restrict__ attn_out) {
  __shared__ float Pf[SEQ][SEQ];
  __shared__ __align__(32) bf16 Pb[SEQ][SPAD];

  const int bh = blockIdx.x;
  const int b = bh >> 3, h = bh & 7;
  const int w = threadIdx.x >> 5;
  const int lane = threadIdx.x & 31, hf = lane >> 4, lr = lane & 15;

  const bf16* qp = qb + (size_t)bh * SEQ * HDIM;
  const bf16* kp = kb + (size_t)bh * SEQ * HDIM;
  const bf16* vp = vtb + (size_t)bh * HDIM * SPAD;

  // ---- S = scale * Q K^T  (K-dim = HDIM = 4 x 32) ----
  v16bf aq[4];
#pragma unroll
  for (int kc = 0; kc < 4; ++kc) {
    const bf16* ap = qp + (size_t)(w * 16 + lr) * HDIM + kc * 32 + 8 * hf;
    aq[kc] = cat8(*(const v8bf*)ap, *(const v8bf*)(ap + 16));
  }
  v8f sacc[5] = {};
#pragma unroll
  for (int nt = 0; nt < 5; ++nt)
#pragma unroll
    for (int kc = 0; kc < 4; ++kc) {
      const bf16* bp = kp + (size_t)(nt * 16 + lr) * HDIM + kc * 32 + 8 * hf;
      sacc[nt] = wmma_bf16(aq[kc], cat8(*(const v8bf*)bp, *(const v8bf*)(bp + 16)), sacc[nt]);
    }
  const float scale = 0.03125f;  // DIM^-0.5 (full emb scale, per reference)
#pragma unroll
  for (int nt = 0; nt < 5; ++nt)
#pragma unroll
    for (int r = 0; r < 8; ++r)
      Pf[w * 16 + 8 * hf + r][nt * 16 + lr] = sacc[nt][r] * scale;
  __syncthreads();

  // ---- softmax over keys, then add relative bias AFTER softmax ----
  const int row = threadIdx.x;
  if (row < SEQ) {
    float mx = -3.4e38f;
    for (int j = 0; j < SEQ; ++j) mx = fmaxf(mx, Pf[row][j]);
    float sum = 0.f;
    for (int j = 0; j < SEQ; ++j) {
      float e = expf(Pf[row][j] - mx);
      Pf[row][j] = e;
      sum += e;
    }
    float is = 1.f / sum;
    for (int j = 0; j < SEQ; ++j) {
      float p = Pf[row][j] * is +
                bias_table[(size_t)(row - j + SEQ - 1) * HEADS + h];
      Pb[row][j] = (bf16)p;
    }
#pragma unroll
    for (int j = SEQ; j < SPAD; ++j) Pb[row][j] = (bf16)0.f;
  }
  __syncthreads();

  // ---- O = P @ V  (K-dim = SPAD = 3 x 32, zero padded) ----
  v8f oacc[8] = {};
#pragma unroll
  for (int kc = 0; kc < 3; ++kc) {
    const bf16* ap = &Pb[w * 16 + lr][kc * 32 + 8 * hf];
    v16bf af = cat8(*(const v8bf*)ap, *(const v8bf*)(ap + 16));
#pragma unroll
    for (int nt = 0; nt < 8; ++nt) {
      const bf16* bp = vp + (size_t)(nt * 16 + lr) * SPAD + kc * 32 + 8 * hf;
      oacc[nt] = wmma_bf16(af, cat8(*(const v8bf*)bp, *(const v8bf*)(bp + 16)), oacc[nt]);
    }
  }
#pragma unroll
  for (int nt = 0; nt < 8; ++nt)
#pragma unroll
    for (int r = 0; r < 8; ++r) {
      int s = w * 16 + 8 * hf + r;
      int d = nt * 16 + lr;
      attn_out[((size_t)b * SEQ + s) * DIM + h * HDIM + d] = oacc[nt][r];
    }
}

// ---------------------------------------------------------------------------
// to_out LayerNorm + residual + ln1  ->  x2 (f32 for residual, bf16 for FC1)
// ---------------------------------------------------------------------------
__global__ __launch_bounds__(256) void ln1_kernel(
    const float* __restrict__ x, const float* __restrict__ aout,
    const float* __restrict__ gto, const float* __restrict__ bto,
    const float* __restrict__ g1, const float* __restrict__ b1,
    float* __restrict__ x2f, bf16* __restrict__ x2b) {
  __shared__ float rs[256], rq[256];
  const int t = threadIdx.x;
  const size_t row = blockIdx.x;
  const float* ar = aout + row * DIM;
  const float* xr = x + row * DIM;
  float v[4], s = 0.f, q = 0.f;
#pragma unroll
  for (int i = 0; i < 4; ++i) {
    v[i] = ar[t + 256 * i];
    s += v[i]; q += v[i] * v[i];
  }
  rs[t] = s; rq[t] = q; __syncthreads();
  for (int off = 128; off > 0; off >>= 1) {
    if (t < off) { rs[t] += rs[t + off]; rq[t] += rq[t + off]; }
    __syncthreads();
  }
  float mu = rs[0] * (1.0f / DIM);
  float var = rq[0] * (1.0f / DIM) - mu * mu;
  float inv = rsqrtf(var + 1e-5f);
  __syncthreads();

  float y[4];
  s = 0.f; q = 0.f;
#pragma unroll
  for (int i = 0; i < 4; ++i) {
    int idx = t + 256 * i;
    float o = (v[i] - mu) * inv * gto[idx] + bto[idx];
    y[i] = xr[idx] + o;
    s += y[i]; q += y[i] * y[i];
  }
  rs[t] = s; rq[t] = q; __syncthreads();
  for (int off = 128; off > 0; off >>= 1) {
    if (t < off) { rs[t] += rs[t + off]; rq[t] += rq[t + off]; }
    __syncthreads();
  }
  mu = rs[0] * (1.0f / DIM);
  var = rq[0] * (1.0f / DIM) - mu * mu;
  inv = rsqrtf(var + 1e-5f);
#pragma unroll
  for (int i = 0; i < 4; ++i) {
    int idx = t + 256 * i;
    float xv = (y[i] - mu) * inv * g1[idx] + b1[idx];
    x2f[row * DIM + idx] = xv;
    x2b[row * DIM + idx] = (bf16)xv;
  }
}

// ---------------------------------------------------------------------------
// FC1 + exact GELU  (x2 @ fc1_w + b -> gelu -> bf16), 64x64 per wave
// ---------------------------------------------------------------------------
__global__ __launch_bounds__(256) void gemm_fc1_kernel(
    const bf16* __restrict__ x2b, const bf16* __restrict__ w1T,
    const float* __restrict__ fc1_b, bf16* __restrict__ hb) {
  const int w = threadIdx.x >> 5;
  const int wm = w & 1, wn = w >> 1;
  const int m0 = blockIdx.x * 128 + wm * 64;
  const int n0 = blockIdx.y * 256 + wn * 64;
  v8f acc[4][4] = {};
  wave_gemm<4, 4>(x2b, DIM, w1T, DIM, DIM, m0, n0, acc);
  const int lane = threadIdx.x & 31, hf = lane >> 4, lr = lane & 15;
#pragma unroll
  for (int mi = 0; mi < 4; ++mi)
#pragma unroll
    for (int ni = 0; ni < 4; ++ni)
#pragma unroll
      for (int r = 0; r < 8; ++r) {
        int m = m0 + mi * 16 + 8 * hf + r;
        int n = n0 + ni * 16 + lr;
        float v = acc[mi][ni][r] + fc1_b[n];
        v = 0.5f * v * (1.0f + erff(v * 0.70710678118654752f));  // exact GELU
        hb[(size_t)m * HID + n] = (bf16)v;
      }
}

// ---------------------------------------------------------------------------
// FC2 + bias + residual  (h @ fc2_w + b + x2 -> f32)
// B tile double-buffered in LDS via async global->LDS copies when available.
// ---------------------------------------------------------------------------
#ifdef HAVE_ASYNC_LDS
__global__ __launch_bounds__(256) void gemm_fc2_kernel(
    const bf16* __restrict__ hb, const bf16* __restrict__ w2T,
    const float* __restrict__ fc2_b, const float* __restrict__ x2f,
    float* __restrict__ fff) {
  __shared__ __align__(64) bf16 btile[2][128][32];  // 2 x 8 KB
  const int w = threadIdx.x >> 5;
  const int wm = w & 1, wn = w >> 1;
  const int m0 = blockIdx.x * 128 + wm * 64;
  const int nblk = blockIdx.y * 128;
  const int n0 = nblk + wn * 32;
  const int t = threadIdx.x;
  const int lane = threadIdx.x & 31, hf = lane >> 4, lr = lane & 15;

  // Each thread async-copies two 16B chunks of the 128x32 bf16 B tile.
  const int e0 = t * 8;           // bf16 element index in tile, chunk 0
  const int e1 = e0 + 2048;       // chunk 1
  auto issue = [&](int k0, int buf) {
    const bf16* g0 = w2T + (size_t)(nblk + (e0 >> 5)) * HID + k0 + (e0 & 31);
    const bf16* g1 = w2T + (size_t)(nblk + (e1 >> 5)) * HID + k0 + (e1 & 31);
    __builtin_amdgcn_global_load_async_to_lds_b128(
        (AS1 v4i*)(AS1 void*)g0,
        (AS3 v4i*)(AS3 void*)&btile[buf][e0 >> 5][e0 & 31],
        0, 0);
    __builtin_amdgcn_global_load_async_to_lds_b128(
        (AS1 v4i*)(AS1 void*)g1,
        (AS3 v4i*)(AS3 void*)&btile[buf][e1 >> 5][e1 & 31],
        0, 0);
  };

  issue(0, 0);
  v8f acc[4][2] = {};
  const int nl = wn * 32;  // wave's n offset inside the block tile
  for (int k0 = 0; k0 < HID; k0 += 32) {
    const int cur = (k0 >> 5) & 1;
    if (k0 + 32 < HID) {
      issue(k0 + 32, cur ^ 1);                  // prefetch next stage
      __builtin_amdgcn_s_wait_asynccnt(2);      // wait for current stage only
    } else {
      __builtin_amdgcn_s_wait_asynccnt(0);
    }
    __syncthreads();  // all waves' async copies for this stage have landed

    v16bf a[4];
#pragma unroll
    for (int mi = 0; mi < 4; ++mi) {
      const bf16* ap = hb + (size_t)(m0 + mi * 16 + lr) * HID + k0 + 8 * hf;
      a[mi] = cat8(*(const v8bf*)ap, *(const v8bf*)(ap + 16));
    }
    __builtin_prefetch(hb + (size_t)(m0 + lr) * HID + k0 + 32, 0, 1);
    v16bf b[2];
#pragma unroll
    for (int ni = 0; ni < 2; ++ni) {
      const bf16* bp = &btile[cur][nl + ni * 16 + lr][8 * hf];
      b[ni] = cat8(*(const v8bf*)bp, *(const v8bf*)(bp + 16));
    }
#pragma unroll
    for (int mi = 0; mi < 4; ++mi)
#pragma unroll
      for (int ni = 0; ni < 2; ++ni)
        acc[mi][ni] = wmma_bf16(a[mi], b[ni], acc[mi][ni]);
    __syncthreads();  // reads done before this buffer is overwritten again
  }

#pragma unroll
  for (int mi = 0; mi < 4; ++mi)
#pragma unroll
    for (int ni = 0; ni < 2; ++ni)
#pragma unroll
      for (int r = 0; r < 8; ++r) {
        int m = m0 + mi * 16 + 8 * hf + r;
        int n = n0 + ni * 16 + lr;
        fff[(size_t)m * DIM + n] =
            acc[mi][ni][r] + fc2_b[n] + x2f[(size_t)m * DIM + n];
      }
}
#else
__global__ __launch_bounds__(256) void gemm_fc2_kernel(
    const bf16* __restrict__ hb, const bf16* __restrict__ w2T,
    const float* __restrict__ fc2_b, const float* __restrict__ x2f,
    float* __restrict__ fff) {
  const int w = threadIdx.x >> 5;
  const int wm = w & 1, wn = w >> 1;
  const int m0 = blockIdx.x * 128 + wm * 64;
  const int n0 = blockIdx.y * 128 + wn * 32;
  v8f acc[4][2] = {};
  wave_gemm<4, 2>(hb, HID, w2T, HID, HID, m0, n0, acc);
  const int lane = threadIdx.x & 31, hf = lane >> 4, lr = lane & 15;
#pragma unroll
  for (int mi = 0; mi < 4; ++mi)
#pragma unroll
    for (int ni = 0; ni < 2; ++ni)
#pragma unroll
      for (int r = 0; r < 8; ++r) {
        int m = m0 + mi * 16 + 8 * hf + r;
        int n = n0 + ni * 16 + lr;
        fff[(size_t)m * DIM + n] =
            acc[mi][ni][r] + fc2_b[n] + x2f[(size_t)m * DIM + n];
      }
}
#endif

// ---------------------------------------------------------------------------
// Final LayerNorm (ln2) -> output
// ---------------------------------------------------------------------------
__global__ __launch_bounds__(256) void ln2_kernel(const float* __restrict__ fff,
                                                  const float* __restrict__ g,
                                                  const float* __restrict__ b,
                                                  float* __restrict__ out) {
  __shared__ float rs[256], rq[256];
  const int t = threadIdx.x;
  const size_t row = blockIdx.x;
  const float* fr = fff + row * DIM;
  float v[4], s = 0.f, q = 0.f;
#pragma unroll
  for (int i = 0; i < 4; ++i) {
    v[i] = fr[t + 256 * i];
    s += v[i]; q += v[i] * v[i];
  }
  rs[t] = s; rq[t] = q; __syncthreads();
  for (int off = 128; off > 0; off >>= 1) {
    if (t < off) { rs[t] += rs[t + off]; rq[t] += rq[t + off]; }
    __syncthreads();
  }
  float mu = rs[0] * (1.0f / DIM);
  float var = rq[0] * (1.0f / DIM) - mu * mu;
  float inv = rsqrtf(var + 1e-5f);
#pragma unroll
  for (int i = 0; i < 4; ++i) {
    int idx = t + 256 * i;
    out[row * DIM + idx] = (v[i] - mu) * inv * g[idx] + b[idx];
  }
}

// ---------------------------------------------------------------------------
extern "C" void kernel_launch(void* const* d_in, const int* in_sizes, int n_in,
                              void* d_out, int out_size, void* d_ws,
                              size_t ws_size, hipStream_t stream) {
  (void)in_sizes; (void)n_in; (void)out_size; (void)ws_size;
  const float* x = (const float*)d_in[0];
  const float* Wq = (const float*)d_in[1];
  const float* Wk = (const float*)d_in[2];
  const float* Wv = (const float*)d_in[3];
  const float* bias_table = (const float*)d_in[4];
  const float* to_out_g = (const float*)d_in[5];
  const float* to_out_b = (const float*)d_in[6];
  const float* ln1_g = (const float*)d_in[7];
  const float* ln1_b = (const float*)d_in[8];
  const float* ln2_g = (const float*)d_in[9];
  const float* ln2_b = (const float*)d_in[10];
  const float* fc1_w = (const float*)d_in[11];
  const float* fc1_b = (const float*)d_in[12];
  const float* fc2_w = (const float*)d_in[13];
  const float* fc2_b = (const float*)d_in[14];
  float* out = (float*)d_out;

  char* ws = (char*)d_ws;
  size_t off = 0;
  auto take = [&](size_t bytes) -> char* {
    char* p = ws + off;
    off += (bytes + 255) & ~(size_t)255;
    return p;
  };
  bf16* xb    = (bf16*)take((size_t)NTOK * DIM * 2);
  bf16* wqkvT = (bf16*)take((size_t)3 * DIM * DIM * 2);
  bf16* w1T   = (bf16*)take((size_t)HID * DIM * 2);
  bf16* w2T   = (bf16*)take((size_t)DIM * HID * 2);
  bf16* qbuf  = (bf16*)take((size_t)NBH * SEQ * HDIM * 2);
  bf16* kbuf  = (bf16*)take((size_t)NBH * SEQ * HDIM * 2);
  bf16* vtb   = (bf16*)take((size_t)NBH * HDIM * SPAD * 2);
  float* aout = (float*)take((size_t)NTOK * DIM * 4);
  float* x2f  = (float*)take((size_t)NTOK * DIM * 4);
  bf16* x2b   = (bf16*)take((size_t)NTOK * DIM * 2);
  bf16* hb    = (bf16*)take((size_t)NTOK * HID * 2);
  float* fff  = (float*)take((size_t)NTOK * DIM * 4);

  // prep: casts / transposes / zero-fill of padded V^T
  {
    size_t n = (size_t)NTOK * DIM;
    cast_kernel<<<(unsigned)((n + 255) / 256), 256, 0, stream>>>(x, xb, n);
  }
  {
    size_t n = (size_t)DIM * DIM;
    unsigned g = (unsigned)((n + 255) / 256);
    transpose_cast_kernel<<<g, 256, 0, stream>>>(Wq, wqkvT, DIM, DIM);
    transpose_cast_kernel<<<g, 256, 0, stream>>>(Wk, wqkvT + (size_t)DIM * DIM, DIM, DIM);
    transpose_cast_kernel<<<g, 256, 0, stream>>>(Wv, wqkvT + (size_t)2 * DIM * DIM, DIM, DIM);
  }
  {
    size_t n = (size_t)DIM * HID;
    unsigned g = (unsigned)((n + 255) / 256);
    transpose_cast_kernel<<<g, 256, 0, stream>>>(fc1_w, w1T, DIM, HID);
    transpose_cast_kernel<<<g, 256, 0, stream>>>(fc2_w, w2T, HID, DIM);
  }
  {
    size_t n16 = (size_t)NBH * HDIM * SPAD * 2 / 16;
    zero_kernel<<<(unsigned)((n16 + 255) / 256), 256, 0, stream>>>((uint4*)vtb, n16);
  }

  // main pipeline
  gemm_qkv_kernel<<<dim3(NTOK / 128, (3 * DIM) / 256), 256, 0, stream>>>(
      xb, wqkvT, qbuf, kbuf, vtb);
  attn_kernel<<<NBH, 160, 0, stream>>>(qbuf, kbuf, vtb, bias_table, aout);
  ln1_kernel<<<NTOK, 256, 0, stream>>>(x, aout, to_out_g, to_out_b, ln1_g,
                                       ln1_b, x2f, x2b);
  gemm_fc1_kernel<<<dim3(NTOK / 128, HID / 256), 256, 0, stream>>>(x2b, w1T,
                                                                   fc1_b, hb);
  gemm_fc2_kernel<<<dim3(NTOK / 128, DIM / 128), 256, 0, stream>>>(hb, w2T,
                                                                   fc2_b, x2f, fff);
  ln2_kernel<<<NTOK, 256, 0, stream>>>(fff, ln2_g, ln2_b, out);
}
